// TrSFRNNAgent_86457691668994
// MI455X (gfx1250) — matrix operand order repaired and checked by reading, not compiled, not run on hardware
//
#include <hip/hip_runtime.h>
#include <stdint.h>

// ---------------------------------------------------------------------------
// Types for CDNA5 WMMA (wave32)
// ---------------------------------------------------------------------------
typedef __attribute__((ext_vector_type(16))) __bf16 v16bf;
typedef __attribute__((ext_vector_type(8)))  float  v8f;
typedef __attribute__((ext_vector_type(8)))  __bf16 bf16x8;

#define TILE_M 128
#define TILE_N 64
#define TILE_K 32

// ---------------------------------------------------------------------------
// Generic bf16 WMMA GEMM:  C = act(A(MxK,bf16) @ B(KxN,bf16) + bias)
// block = 128 threads (4 waves). Block tile 128x64; wave w computes rows
// [w*32, w*32+32) as 2 A-fragments x 4 N-tiles = eight v_wmma_f32_16x16x32_bf16
// per K-step. A tile is staged with GLOBAL_LOAD_ASYNC_TO_LDS_B128 (ASYNCcnt),
// B tile is staged+transposed through VGPRs so WMMA fragment reads are
// contiguous ds_load_b128 pairs.
// act: 0=none, 1=leaky_relu(0.01), 2=relu.  Cf (fp32) and/or Cb (bf16) out.
// Requires: M%128==0, N%64==0, K%32==0 (true for every call below).
// ---------------------------------------------------------------------------
__global__ __launch_bounds__(128)
void gemm_bf16_wmma(const __bf16* __restrict__ A, const __bf16* __restrict__ B,
                    const float* __restrict__ bias,
                    float* __restrict__ Cf, __bf16* __restrict__ Cb,
                    int M, int N, int K, int act)
{
    __shared__ __align__(16) __bf16 Asm[TILE_M * TILE_K];  // 8KB  [row][k], stride 32
    __shared__ __align__(16) __bf16 Bsm[TILE_N * TILE_K];  // 4KB  [col][k] (transposed)

    const int t    = threadIdx.x;
    const int wave = t >> 5;
    const int lane = t & 31;
    const int lh   = lane >> 4;   // 0/1: K-half selector per ISA operand layout
    const int ll   = lane & 15;
    const int m0   = blockIdx.x * TILE_M;
    const int n0   = blockIdx.y * TILE_N;

    const unsigned a_lds_base = (unsigned)(unsigned long long)(const void*)&Asm[0];

    v8f acc[2][4] = {};

    for (int k0 = 0; k0 < K; k0 += TILE_K) {
        if (k0 + TILE_K < K) {  // prefetch next B tile -> global_prefetch_b8
            __builtin_prefetch(&B[(size_t)(k0 + TILE_K + (t >> 3)) * N + n0], 0, 1);
        }
        // ---- A tile 128x32 bf16 = 512 chunks of 16B: async copy to LDS ----
        // chunk c: row = c>>2, k = (c&3)*8 ; LDS byte offset = c*16
        #pragma unroll
        for (int c = t; c < 512; c += 128) {
            unsigned lds = a_lds_base + (unsigned)c * 16u;
            unsigned long long ga = (unsigned long long)(const void*)
                (&A[(size_t)(m0 + (c >> 2)) * K + k0 + (c & 3) * 8]);
            asm volatile("global_load_async_to_lds_b128 %0, %1, off"
                         :: "v"(lds), "v"(ga) : "memory");
        }
        // ---- B tile 32x64 -> transposed into [col][k] via VGPRs ----
        #pragma unroll
        for (int c = t; c < 256; c += 128) {
            int k = c >> 3, nc = (c & 7) * 8;
            bf16x8 v = *(const bf16x8*)(&B[(size_t)(k0 + k) * N + n0 + nc]);
            #pragma unroll
            for (int i = 0; i < 8; ++i) Bsm[(nc + i) * TILE_K + k] = v[i];
        }
        asm volatile("s_wait_asynccnt 0x0" ::: "memory");
        __syncthreads();

        // A fragments per ISA: lane<16 -> K {0..7,16..23}, lane>=16 -> K {8..15,24..31}
        v16bf af[2];
        #pragma unroll
        for (int mf = 0; mf < 2; ++mf) {
            const __bf16* ap = &Asm[(wave * 32 + mf * 16 + ll) * TILE_K + lh * 8];
            #pragma unroll
            for (int i = 0; i < 8; ++i) { af[mf][i] = ap[i]; af[mf][8 + i] = ap[16 + i]; }
        }
        #pragma unroll
        for (int nt = 0; nt < 4; ++nt) {
            // B fragment: lane<16 -> K {0..7,16..23} of col ll, lane>=16 -> K {8..15,24..31}
            const __bf16* bp = &Bsm[(nt * 16 + ll) * TILE_K + lh * 8];
            v16bf bfrag;
            #pragma unroll
            for (int i = 0; i < 8; ++i) { bfrag[i] = bp[i]; bfrag[8 + i] = bp[16 + i]; }
            acc[0][nt] = __builtin_amdgcn_wmma_f32_16x16x32_bf16(
                false, af[0], false, bfrag, (short)0, acc[0][nt], false, false);
            acc[1][nt] = __builtin_amdgcn_wmma_f32_16x16x32_bf16(
                false, af[1], false, bfrag, (short)0, acc[1][nt], false, false);
        }
        __syncthreads();
    }

    // Writeback per ISA C layout: VGPR r, lane l -> M = 8*(l>>4)+r, N = l&15
    #pragma unroll
    for (int mf = 0; mf < 2; ++mf) {
        #pragma unroll
        for (int nt = 0; nt < 4; ++nt) {
            int col = n0 + nt * 16 + ll;
            float bv = bias ? bias[col] : 0.0f;
            #pragma unroll
            for (int r = 0; r < 8; ++r) {
                int row = m0 + wave * 32 + mf * 16 + lh * 8 + r;
                float v = acc[mf][nt][r] + bv;
                if (act == 1)      v = v > 0.f ? v : 0.01f * v;
                else if (act == 2) v = v > 0.f ? v : 0.f;
                if (Cf) Cf[(size_t)row * N + col] = v;
                if (Cb) Cb[(size_t)row * N + col] = (__bf16)v;
            }
        }
    }
}

// ---------------------------------------------------------------------------
// fp32 -> bf16 conversion
// ---------------------------------------------------------------------------
__global__ void cvt_bf16_k(const float* __restrict__ src, __bf16* __restrict__ dst, int n)
{
    int i = blockIdx.x * blockDim.x + threadIdx.x;
    if (i < n) dst[i] = (__bf16)src[i];
}

// ---------------------------------------------------------------------------
// Preprocess: own_obs, enemy_feats, query/own_value GEMVs, ally & enemy
// attention. One block (64 threads) per BSN row. Writes x (1024x192 fp32)
// and ef_in (8192x17 fp32).
// ---------------------------------------------------------------------------
__global__ __launch_bounds__(64)
void preprocess_k(const float* __restrict__ in,  // 1024 x 294
                  const float* Wq,  const float* bq,
                  const float* Wov, const float* bov,
                  const float* Wak, const float* bak,
                  const float* Wav, const float* bav,
                  const float* Wek, const float* bek,
                  const float* Wev, const float* bev,
                  float* __restrict__ x, float* __restrict__ ef_in)
{
    __shared__ float oo[47];
    __shared__ float qs[64];
    __shared__ float ks[8 * 64];
    __shared__ float vs[8 * 64];
    __shared__ float sc[8];

    const int r = blockIdx.x;
    const int t = threadIdx.x;
    const float* row = in + (size_t)r * 294;

    if (t < 47) {
        float v;
        if (t < 32) v = row[t];
        else if (t < 40) { int agent = (r & 7) + 1; v = (float)((agent >> (t - 32)) & 1); }
        else if (t < 46) v = row[272 + (t - 40)];
        else { float s = 0.f; for (int e = 0; e < 8; ++e) s += row[272 + 6 + e]; v = s; }
        oo[t] = v;
    }
    for (int idx = t; idx < 136; idx += 64) {
        int e = idx / 17, f = idx - e * 17;
        float v = (f < 16) ? row[32 + e * 16 + f] : row[272 + 6 + e];
        ef_in[((size_t)r * 8 + e) * 17 + f] = v;
    }
    __syncthreads();

    // q and own_f (47 -> 64)
    {
        float aq = bq[t], ao = bov[t];
        for (int i = 0; i < 47; ++i) {
            float o = oo[i];
            aq += o * Wq[i * 64 + t];
            ao += o * Wov[i * 64 + t];
        }
        qs[t] = aq;
        x[(size_t)r * 192 + t] = ao;
    }
    __syncthreads();

    // ally attention: 7 entities x 16 feats (obs offset 160)
    for (int e = 0; e < 7; ++e) {
        float ak = bak[t], av = bav[t];
        const float* af = row + 160 + e * 16;
        for (int i = 0; i < 16; ++i) {
            float o = af[i];
            ak += o * Wak[i * 64 + t];
            av += o * Wav[i * 64 + t];
        }
        ks[e * 64 + t] = ak; vs[e * 64 + t] = av;
    }
    __syncthreads();
    if (t < 7) {
        float d = 0.f;
        for (int i = 0; i < 64; ++i) d += qs[i] * ks[t * 64 + i];
        sc[t] = d * 0.125f;
    }
    __syncthreads();
    {
        float m = sc[0];
        for (int e = 1; e < 7; ++e) m = fmaxf(m, sc[e]);
        float p[7], den = 0.f;
        for (int e = 0; e < 7; ++e) { p[e] = __expf(sc[e] - m); den += p[e]; }
        float invd = 1.0f / den, a = 0.f;
        for (int e = 0; e < 7; ++e) a += p[e] * invd * vs[e * 64 + t];
        x[(size_t)r * 192 + 64 + t] = a;
    }
    __syncthreads();

    // enemy attention: 8 entities x 17 feats (16 obs + atk action)
    for (int e = 0; e < 8; ++e) {
        float ek = bek[t], ev = bev[t];
        const float* nf = row + 32 + e * 16;
        for (int i = 0; i < 16; ++i) {
            float o = nf[i];
            ek += o * Wek[i * 64 + t];
            ev += o * Wev[i * 64 + t];
        }
        float atk = row[272 + 6 + e];
        ek += atk * Wek[16 * 64 + t];
        ev += atk * Wev[16 * 64 + t];
        ks[e * 64 + t] = ek; vs[e * 64 + t] = ev;
    }
    __syncthreads();
    if (t < 8) {
        float d = 0.f;
        for (int i = 0; i < 64; ++i) d += qs[i] * ks[t * 64 + i];
        sc[t] = d * 0.125f;
    }
    __syncthreads();
    {
        float m = sc[0];
        for (int e = 1; e < 8; ++e) m = fmaxf(m, sc[e]);
        float p[8], den = 0.f;
        for (int e = 0; e < 8; ++e) { p[e] = __expf(sc[e] - m); den += p[e]; }
        float invd = 1.0f / den, a = 0.f;
        for (int e = 0; e < 8; ++e) a += p[e] * invd * vs[e * 64 + t];
        x[(size_t)r * 192 + 128 + t] = a;
    }
}

// ---------------------------------------------------------------------------
// Task encoder (batch 1): te = leaky(tw @ W1 + b1) @ W2 + b2  -> 256 floats
// ---------------------------------------------------------------------------
__global__ __launch_bounds__(256)
void task_enc_k(const float* tw_in, const float* eps,
                const float* W1, const float* b1,
                const float* W2, const float* b2, float* te)
{
    __shared__ float tw[16];
    __shared__ float t1[1024];
    int t = threadIdx.x;
    if (t < 16) {
        float e = eps[t];
        e = fminf(0.5f, fmaxf(-0.5f, e));
        tw[t] = tw_in[t] + e * 0.1f;
    }
    __syncthreads();
    for (int k = t; k < 1024; k += 256) {
        float a = b1[k];
        for (int i = 0; i < 16; ++i) a += tw[i] * W1[i * 1024 + k];
        t1[k] = a > 0.f ? a : 0.01f * a;
    }
    __syncthreads();
    {
        float a = b2[t];
        for (int k = 0; k < 1024; ++k) a += t1[k] * W2[k * 256 + t];
        te[t] = a;
    }
}

// ---------------------------------------------------------------------------
// GRU gate fusion; writes h (fp32 output) and bf16 copy into A1[:, :256]
// ---------------------------------------------------------------------------
__global__ void gru_elem_k(const float* __restrict__ gx, const float* __restrict__ gh,
                           const float* __restrict__ h0,
                           float* __restrict__ hout, __bf16* __restrict__ A1)
{
    int idx = blockIdx.x * blockDim.x + threadIdx.x;  // 1024*256
    if (idx >= 1024 * 256) return;
    int r = idx >> 8, j = idx & 255;
    const float* gxr = gx + (size_t)r * 768;
    const float* ghr = gh + (size_t)r * 768;
    float xr = gxr[j], xz = gxr[256 + j], xn = gxr[512 + j];
    float hr = ghr[j], hz = ghr[256 + j], hn = ghr[512 + j];
    float rg = 1.f / (1.f + __expf(-(xr + hr)));
    float z  = 1.f / (1.f + __expf(-(xz + hz)));
    float n  = tanhf(xn + rg * hn);
    float h  = (1.f - z) * n + z * h0[idx];
    hout[idx] = h;
    A1[(size_t)r * 512 + j] = (__bf16)h;
}

__global__ void fill_te_k(const float* __restrict__ te, __bf16* __restrict__ A1)
{
    int idx = blockIdx.x * blockDim.x + threadIdx.x;  // 1024*256
    if (idx >= 1024 * 256) return;
    int r = idx >> 8, j = idx & 255;
    A1[(size_t)r * 512 + 256 + j] = (__bf16)te[j];
}

// een1: relu(ef_in(8192x17) @ W(17x256) + b) -> bf16
__global__ void een1_k(const float* __restrict__ ef_in, const float* __restrict__ W,
                       const float* __restrict__ b, __bf16* __restrict__ out)
{
    int idx = blockIdx.x * blockDim.x + threadIdx.x;  // 8192*256
    if (idx >= 8192 * 256) return;
    int row = idx >> 8, j = idx & 255;
    const float* fr = ef_in + (size_t)row * 17;
    float a = b[j];
    for (int i = 0; i < 17; ++i) a += fr[i] * W[i * 256 + j];
    out[idx] = (__bf16)(a > 0.f ? a : 0.f);
}

// wo: relu(psi_hid) @ W(128x6) + b -> psi columns 0..5
__global__ void wo_k(const float* __restrict__ psi_hid, const float* __restrict__ W,
                     const float* __restrict__ b, float* __restrict__ out_psi)
{
    int idx = blockIdx.x * blockDim.x + threadIdx.x;  // 1024*16*6
    if (idx >= 1024 * 16 * 6) return;
    int j = idx % 6, rp = idx / 6;
    const float* ph = psi_hid + (size_t)rp * 128;
    float a = b[j];
    for (int i = 0; i < 128; ++i) {
        float v = ph[i];
        v = v > 0.f ? v : 0.f;
        a += v * W[i * 6 + j];
    }
    out_psi[(size_t)rp * 14 + j] = a;
}

// atk combine: sum_j relu(efA[b,e,j] + psA[b,p,j]) * w[j] + bias -> psi cols 6..13
// One wave32 per (b,p,e) triple.
__global__ __launch_bounds__(256)
void atk_combine_k(const float* __restrict__ efA, const float* __restrict__ psA,
                   const float* __restrict__ w, const float* __restrict__ bscal,
                   float* __restrict__ out_psi)
{
    int gwave = (blockIdx.x * blockDim.x + threadIdx.x) >> 5;  // 131072 waves
    int lane  = threadIdx.x & 31;
    int b = gwave >> 7;
    int p = (gwave >> 3) & 15;
    int e = gwave & 7;
    const float* ea = efA + ((size_t)b * 8 + e) * 256;
    const float* pa = psA + ((size_t)b * 16 + p) * 256;
    float acc = 0.f;
    for (int i = lane; i < 256; i += 32) {
        float v = ea[i] + pa[i];
        v = v > 0.f ? v : 0.f;
        acc += v * w[i];
    }
    for (int off = 16; off > 0; off >>= 1) acc += __shfl_down(acc, off, 32);
    if (lane == 0) out_psi[((size_t)b * 16 + p) * 14 + 6 + e] = acc + bscal[0];
}

// ---------------------------------------------------------------------------
// Host launcher
// ---------------------------------------------------------------------------
extern "C" void kernel_launch(void* const* d_in, const int* in_sizes, int n_in,
                              void* d_out, int out_size, void* d_ws, size_t ws_size,
                              hipStream_t stream)
{
    // d_in layout (setup_inputs dict order, params tuples flattened W,b):
    const float* inputs = (const float*)d_in[0];
    const float* h0     = (const float*)d_in[2];
    const float* task_w = (const float*)d_in[3];
    const float* eps    = (const float*)d_in[4];
    const float* Wq   = (const float*)d_in[5];  const float* bq   = (const float*)d_in[6];
    const float* Wov  = (const float*)d_in[7];  const float* bov  = (const float*)d_in[8];
    const float* Wak  = (const float*)d_in[9];  const float* bak  = (const float*)d_in[10];
    const float* Wav  = (const float*)d_in[11]; const float* bav  = (const float*)d_in[12];
    const float* Wek  = (const float*)d_in[13]; const float* bek  = (const float*)d_in[14];
    const float* Wev  = (const float*)d_in[15]; const float* bev  = (const float*)d_in[16];
    const float* wih  = (const float*)d_in[17];
    const float* whh  = (const float*)d_in[18];
    const float* bih  = (const float*)d_in[19];
    const float* bhh  = (const float*)d_in[20];
    const float* te1W = (const float*)d_in[21]; const float* te1b = (const float*)d_in[22];
    const float* te2W = (const float*)d_in[23]; const float* te2b = (const float*)d_in[24];
    const float* ps1W = (const float*)d_in[25]; const float* ps1b = (const float*)d_in[26];
    const float* ps2W = (const float*)d_in[27]; const float* ps2b = (const float*)d_in[28];
    const float* woW  = (const float*)d_in[29]; const float* wob  = (const float*)d_in[30];
    const float* atk1W = (const float*)d_in[31]; const float* atk1b = (const float*)d_in[32];
    const float* atk2W = (const float*)d_in[33]; const float* atk2b = (const float*)d_in[34];
    const float* een1W = (const float*)d_in[35]; const float* een1b = (const float*)d_in[36];
    const float* een2W = (const float*)d_in[37]; const float* een2b = (const float*)d_in[38];

    float* out_h   = (float*)d_out;            // 1024*256
    float* out_psi = (float*)d_out + 262144;   // 1024*16*14

    // Workspace carving (256B aligned)
    uint8_t* base = (uint8_t*)d_ws;
    size_t off = 0;
    auto carve = [&](size_t bytes) -> void* {
        void* p = base + off;
        off = (off + bytes + 255) & ~(size_t)255;
        return p;
    };
    float*  x_f        = (float*)carve(1024 * 192 * 4);
    float*  ef_in      = (float*)carve(8192 * 17 * 4);
    __bf16* x_bf       = (__bf16*)carve(1024 * 192 * 2);
    __bf16* h0_bf      = (__bf16*)carve(1024 * 256 * 2);
    __bf16* wih_bf     = (__bf16*)carve(192 * 768 * 2);
    __bf16* whh_bf     = (__bf16*)carve(256 * 768 * 2);
    __bf16* ps1W_bf    = (__bf16*)carve(512 * 1024 * 2);
    __bf16* ps2W_bf    = (__bf16*)carve(1024 * 2048 * 2);
    __bf16* een2W_bf   = (__bf16*)carve(256 * 256 * 2);
    __bf16* atk1W_bf   = (__bf16*)carve(384 * 256 * 2);
    float*  gx         = (float*)carve((size_t)1024 * 768 * 4);
    float*  gh         = (float*)carve((size_t)1024 * 768 * 4);
    float*  te         = (float*)carve(256 * 4);
    __bf16* A1         = (__bf16*)carve(1024 * 512 * 2);
    __bf16* A2         = (__bf16*)carve((size_t)1024 * 1024 * 2);
    float*  psi_hid    = (float*)carve((size_t)1024 * 2048 * 4);
    __bf16* psi_hid_bf = (__bf16*)carve((size_t)1024 * 2048 * 2);
    __bf16* ef1_bf     = (__bf16*)carve((size_t)8192 * 256 * 2);
    __bf16* ef_bf      = (__bf16*)carve((size_t)8192 * 256 * 2);
    float*  efA        = (float*)carve((size_t)8192 * 256 * 4);
    float*  psA        = (float*)carve((size_t)16384 * 256 * 4);
    (void)ws_size; (void)in_sizes; (void)n_in; (void)out_size;

    auto cvt = [&](const float* s, __bf16* d, int n) {
        cvt_bf16_k<<<(n + 255) / 256, 256, 0, stream>>>(s, d, n);
    };

    // Weight / state conversions to bf16
    cvt(wih, wih_bf, 192 * 768);
    cvt(whh, whh_bf, 256 * 768);
    cvt(ps1W, ps1W_bf, 512 * 1024);
    cvt(ps2W, ps2W_bf, 1024 * 2048);
    cvt(een2W, een2W_bf, 256 * 256);
    cvt(atk1W, atk1W_bf, 384 * 256);
    cvt(h0, h0_bf, 1024 * 256);

    // Preprocess + attention
    preprocess_k<<<1024, 64, 0, stream>>>(inputs, Wq, bq, Wov, bov, Wak, bak,
                                          Wav, bav, Wek, bek, Wev, bev, x_f, ef_in);
    cvt(x_f, x_bf, 1024 * 192);

    // GRU input/hidden GEMMs (WMMA)
    gemm_bf16_wmma<<<dim3(1024 / 128, 768 / 64), 128, 0, stream>>>(
        x_bf, wih_bf, bih, gx, nullptr, 1024, 768, 192, 0);
    gemm_bf16_wmma<<<dim3(1024 / 128, 768 / 64), 128, 0, stream>>>(
        h0_bf, whh_bf, bhh, gh, nullptr, 1024, 768, 256, 0);

    // Task encoder (batch 1) and GRU fusion
    task_enc_k<<<1, 256, 0, stream>>>(task_w, eps, te1W, te1b, te2W, te2b, te);
    gru_elem_k<<<1024, 256, 0, stream>>>(gx, gh, h0, out_h, A1);
    fill_te_k<<<1024, 256, 0, stream>>>(te, A1);

    // psi1: leaky(A1(1024x512) @ W(512x1024) + b) -> bf16 A2
    gemm_bf16_wmma<<<dim3(8, 16), 128, 0, stream>>>(
        A1, ps1W_bf, ps1b, nullptr, A2, 1024, 1024, 512, 1);
    // psi2: A2(1024x1024) @ W(1024x2048) + b -> psi_hid (f32 + bf16)
    gemm_bf16_wmma<<<dim3(8, 32), 128, 0, stream>>>(
        A2, ps2W_bf, ps2b, psi_hid, psi_hid_bf, 1024, 2048, 1024, 0);

    // wo head -> psi columns 0..5
    wo_k<<<(1024 * 16 * 6 + 255) / 256, 256, 0, stream>>>(psi_hid, woW, wob, out_psi);

    // enemy encoder
    een1_k<<<8192, 256, 0, stream>>>(ef_in, een1W, een1b, ef1_bf);
    gemm_bf16_wmma<<<dim3(64, 4), 128, 0, stream>>>(
        ef1_bf, een2W_bf, een2b, nullptr, ef_bf, 8192, 256, 256, 0);

    // atk1 factorization: ef @ W_top, psi_hid @ W_bot (+bias)
    gemm_bf16_wmma<<<dim3(64, 4), 128, 0, stream>>>(
        ef_bf, atk1W_bf, nullptr, efA, nullptr, 8192, 256, 256, 0);
    gemm_bf16_wmma<<<dim3(128, 4), 128, 0, stream>>>(
        psi_hid_bf, atk1W_bf + 256 * 256, atk1b, psA, nullptr, 16384, 256, 128, 0);

    // relu + atk2 dot -> psi columns 6..13
    atk_combine_k<<<16384, 256, 0, stream>>>(efA, psA, atk2W, atk2b, out_psi);
}
